// ComplexWindowAttention_18640158064974
// MI455X (gfx1250) — matrix operand order repaired and checked by reading, not compiled, not run on hardware
//
#include <hip/hip_runtime.h>
#include <math.h>

// ---------------------------------------------------------------------------
// Complex shifted-window attention, fully fused, one workgroup per window.
//  * Karatsuba complex GEMM: 3 WMMA chains per tile (matrix-op bound kernel).
//  * f16 weight planes {R,I,R+I} pre-converted once into d_ws (L2-resident).
//  * All A-operands stored in LDS pre-swizzled to the WMMA lane layout so
//    every fragment (A and B) is exactly two ds_load_b128 / global b128 pairs
//    per lane -> ~12 DS issues per 3 WMMAs instead of ~30.
// ---------------------------------------------------------------------------

typedef __attribute__((ext_vector_type(16))) _Float16 v16h;
typedef __attribute__((ext_vector_type(8)))  _Float16 v8h;
typedef __attribute__((ext_vector_type(8)))  float    v8f;

namespace cwa {
constexpr int B = 16, H = 64, W = 64, C = 256;
constexpr int SH = 4, SW = 4;
constexpr int NH = 8, N = 64, HD = 32;
constexpr int IMG = B * H * W * C;             // imag-plane offset in d_out
// d_ws f16 plane offsets (elements)
constexpr int WQKV  = 3 * C * C;               // 196608 per qkv plane
constexpr int WPROJ = C * C;                   // 65536 per proj plane
constexpr int OQ_R = 0, OQ_I = WQKV, OQ_S = 2 * WQKV;
constexpr int OP_R = 3 * WQKV, OP_I = OP_R + WPROJ, OP_S = OP_R + 2 * WPROJ;
// LDS offsets (bytes)
constexpr int L_XR = 0,      L_XI = 32768,  L_XS = 65536;    // X planes 64x256 (swz)
constexpr int L_OR = 98304,  L_OI = 131072, L_OS = 163840;   // Out planes 64x256 (swz)
constexpr int L_Q  = 196608;                                  // q planes 64x32 (swz)
constexpr int L_K  = 208896;                                  // k planes 64x32 row-major
constexpr int L_VT = 221184;                                  // vT planes 32x64
constexpr int L_SRE = 233472, L_SIM = 249856;                 // scores f32 64x64
constexpr int L_AF = 266240;                                  // attn planes 64x64 (swz)
constexpr int LDS_BYTES = 290816;                             // 284 KB of 320 KB WGP LDS
}
using namespace cwa;

__device__ __forceinline__ v8f wmma_f16(v16h a, v16h b, v8f c) {
  return __builtin_amdgcn_wmma_f32_16x16x32_f16(false, a, false, b,
                                                (short)0, c, false, false);
}

// Swizzled A-plane address: matrix of 16x32 tiles stored tile-major, within a
// tile 32 lanes x 16 contiguous halves, matching ISA 7.12.2 A-fragment layout.
// element (m,k):  lane = m%16 + 16*((k>>3)&1) ; j = ((k&16)>>1) + (k&7)
__device__ __forceinline__ int aswz(int m, int k, int ktiles) {
  int mt = m >> 4, kt = k >> 5, mm = m & 15, kk = k & 31;
  int lane = mm + (((kk >> 3) & 1) << 4);
  int j = ((kk & 16) >> 1) + (kk & 7);
  return ((mt * ktiles + kt) * 32 + lane) * 16 + j;
}

// A fragment from a swizzled plane: two ds_load_b128 per lane.
__device__ __forceinline__ v16h frag_a_swz(const _Float16* base, int ktiles, int mt, int kt) {
  const int lane = threadIdx.x & 31;
  const v8h* p = reinterpret_cast<const v8h*>(base + ((mt * ktiles + kt) * 32 + lane) * 16);
  v8h lo = p[0], hi = p[1];
  v16h f;
#pragma unroll
  for (int j = 0; j < 8; ++j) { f[j] = lo[j]; f[8 + j] = hi[j]; }
  return f;
}

// B fragment 32x16, element (k,n) = base[n*ld + k]: per lane k = kb..kb+7 and
// kb+16..kb+23 (kb = k0 + 8*half) -> two b128 loads (LDS or global).
__device__ __forceinline__ v16h frag_b_cnk(const _Float16* base, int ld, int k0, int n0) {
  const int lane = threadIdx.x & 31;
  const int n  = n0 + (lane & 15);
  const int kb = k0 + ((lane & 16) ? 8 : 0);
  const v8h* p = reinterpret_cast<const v8h*>(base + n * ld + kb);
  v8h lo = p[0], hi = p[2];
  v16h f;
#pragma unroll
  for (int j = 0; j < 8; ++j) { f[j] = lo[j]; f[8 + j] = hi[j]; }
  return f;
}

// Pack two f32 -> f16x2 and store as one 32-bit LDS write (p must be 2-aligned).
__device__ __forceinline__ void store_h2(_Float16* p, float lo, float hi) {
  _Float16 a = (_Float16)lo, b = (_Float16)hi;
  unsigned int u = (unsigned int)__builtin_bit_cast(unsigned short, a) |
                   ((unsigned int)__builtin_bit_cast(unsigned short, b) << 16);
  *reinterpret_cast<unsigned int*>(p) = u;
}

__device__ __forceinline__ int region9(int h, int w) {
  int rb = (h < 56) ? 0 : (h < 60) ? 1 : 2;
  int cb = (w < 56) ? 0 : (w < 60) ? 1 : 2;
  return rb * 3 + cb;
}

// ---- Pre-pass: convert weights to f16 planes {R, I, R+I} in d_ws ----------
__global__ __launch_bounds__(256)
void cwa_cvt_weights(const float* __restrict__ wqkv_r, const float* __restrict__ wqkv_i,
                     const float* __restrict__ wproj_r, const float* __restrict__ wproj_i,
                     _Float16* __restrict__ ws) {
  int idx = blockIdx.x * 256 + threadIdx.x;
  if (idx < WQKV) {
    float r = wqkv_r[idx], i = wqkv_i[idx];
    ws[OQ_R + idx] = (_Float16)r;
    ws[OQ_I + idx] = (_Float16)i;
    ws[OQ_S + idx] = (_Float16)(r + i);
  }
  if (idx < WPROJ) {
    float r = wproj_r[idx], i = wproj_i[idx];
    ws[OP_R + idx] = (_Float16)r;
    ws[OP_I + idx] = (_Float16)i;
    ws[OP_S + idx] = (_Float16)(r + i);
  }
}

// ---- Fused per-window kernel ----------------------------------------------
__global__ __launch_bounds__(256)
void cwa_fused_kernel(const float* __restrict__ x_real, const float* __restrict__ x_imag,
                      const float* __restrict__ bqkv_r, const float* __restrict__ bqkv_i,
                      const float* __restrict__ bproj_r, const float* __restrict__ bproj_i,
                      const float* __restrict__ rpb_r,  const float* __restrict__ rpb_i,
                      const _Float16* __restrict__ ws,
                      float* __restrict__ out) {
  extern __shared__ char smem[];
  _Float16* Xr  = (_Float16*)(smem + L_XR);     // swizzled, ktiles=8
  _Float16* Xi  = (_Float16*)(smem + L_XI);
  _Float16* Xs  = (_Float16*)(smem + L_XS);
  _Float16* Or  = (_Float16*)(smem + L_OR);     // swizzled, ktiles=8
  _Float16* Oi  = (_Float16*)(smem + L_OI);
  _Float16* Os  = (_Float16*)(smem + L_OS);
  _Float16* qr  = (_Float16*)(smem + L_Q);      // swizzled, ktiles=1
  _Float16* qi  = qr + N * HD;
  _Float16* qs  = qi + N * HD;
  _Float16* kr  = (_Float16*)(smem + L_K);      // row-major [N][HD] (B operand)
  _Float16* ki  = kr + N * HD;
  _Float16* ks  = ki + N * HD;
  _Float16* vtr = (_Float16*)(smem + L_VT);     // transposed [HD][N] (B operand)
  _Float16* vti = vtr + HD * N;
  _Float16* vts = vti + HD * N;
  float*    sre = (float*)(smem + L_SRE);
  float*    sim = (float*)(smem + L_SIM);
  _Float16* afr = (_Float16*)(smem + L_AF);     // swizzled, ktiles=2
  _Float16* afi = afr + N * N;
  _Float16* afs = afi + N * N;

  const int tid  = threadIdx.x;
  const int wid  = tid >> 5;
  const int lane = tid & 31;
  const int blk  = blockIdx.x;
  const int b    = blk >> 6;
  const int wi   = blk & 63;
  const int iw   = wi >> 3;
  const int jw   = wi & 7;

  // ---- Phase 1: gather shifted window, f32 -> swizzled f16 {R,I,R+I} ----
  for (int idx2 = tid; idx2 < (N * C) / 2; idx2 += 256) {
    int n = idx2 >> 7, c = (idx2 & 127) * 2;          // two adjacent channels
    int h = (iw * 8 + (n >> 3) + SH) & 63;
    int w = (jw * 8 + (n & 7) + SW) & 63;
    int g = ((b * H + h) * W + w) * C + c;
    float2 r2 = *reinterpret_cast<const float2*>(x_real + g);
    float2 i2 = *reinterpret_cast<const float2*>(x_imag + g);
    int a0 = aswz(n, c, 8);                           // c even -> j even, pair adjacent
    store_h2(Xr + a0, r2.x, r2.y);
    store_h2(Xi + a0, i2.x, i2.y);
    store_h2(Xs + a0, r2.x + i2.x, r2.y + i2.y);
  }
  __syncthreads();

  const float qscale = 0.17677669529663687f;   // HD^-0.5
  const int mt = wid >> 1, nt = wid & 1;       // wave tile in 64x32 output grid

  for (int hh = 0; hh < NH; ++hh) {
    // ---- (a) q/k/v head slice: Karatsuba (64x256)x(256x32) ----
    for (int mat = 0; mat < 3; ++mat) {
      const int wofs = (mat * C + hh * HD) * C;
      v8f p1 = {}, p2 = {}, p3 = {};
#pragma unroll
      for (int kk = 0; kk < 8; ++kk) {
        int k0 = kk * 32;
        v16h fa_r = frag_a_swz(Xr, 8, mt, kk);
        v16h fa_i = frag_a_swz(Xi, 8, mt, kk);
        v16h fa_s = frag_a_swz(Xs, 8, mt, kk);
        v16h fb_r = frag_b_cnk(ws + OQ_R + wofs, C, k0, nt * 16);
        v16h fb_i = frag_b_cnk(ws + OQ_I + wofs, C, k0, nt * 16);
        v16h fb_s = frag_b_cnk(ws + OQ_S + wofs, C, k0, nt * 16);
        p1 = wmma_f16(fa_r, fb_r, p1);
        p2 = wmma_f16(fa_i, fb_i, p2);
        p3 = wmma_f16(fa_s, fb_s, p3);
      }
      int col  = nt * 16 + (lane & 15);
      int mrow = mt * 16 + ((lane & 16) ? 8 : 0);
      float br = bqkv_r[mat * C + hh * HD + col];
      float bi = bqkv_i[mat * C + hh * HD + col];
      float sc = (mat == 0) ? qscale : 1.0f;
      if (mat == 2) {   // V transposed (B operand of attn*V)
#pragma unroll
        for (int v = 0; v < 8; ++v) {
          float cr = p1[v] - p2[v] + br;
          float ci = p3[v] - p1[v] - p2[v] + bi;
          vtr[col * N + mrow + v] = (_Float16)cr;
          vti[col * N + mrow + v] = (_Float16)ci;
          vts[col * N + mrow + v] = (_Float16)(cr + ci);
        }
      } else if (mat == 1) {  // K row-major (B operand of q*k^T)
#pragma unroll
        for (int v = 0; v < 8; ++v) {
          float cr = p1[v] - p2[v] + br;
          float ci = p3[v] - p1[v] - p2[v] + bi;
          kr[(mrow + v) * HD + col] = (_Float16)cr;
          ki[(mrow + v) * HD + col] = (_Float16)ci;
          ks[(mrow + v) * HD + col] = (_Float16)(cr + ci);
        }
      } else {                // Q swizzled (A operand of q*k^T)
#pragma unroll
        for (int v = 0; v < 8; ++v) {
          float cr = (p1[v] - p2[v] + br) * sc;
          float ci = (p3[v] - p1[v] - p2[v] + bi) * sc;
          int a0 = aswz(mrow + v, col, 1);
          qr[a0] = (_Float16)cr;
          qi[a0] = (_Float16)ci;
          qs[a0] = (_Float16)(cr + ci);
        }
      }
    }
    __syncthreads();

    // ---- (b) scores S = q k^T + rel-pos bias + shift mask ----
#pragma unroll
    for (int t = 0; t < 2; ++t) {
      int p = wid * 2 + t;
      int smt = p >> 2, snt = p & 3;
      v16h qa_r = frag_a_swz(qr, 1, smt, 0);
      v16h qa_i = frag_a_swz(qi, 1, smt, 0);
      v16h qa_s = frag_a_swz(qs, 1, smt, 0);
      v16h kb_r = frag_b_cnk(kr, HD, 0, snt * 16);   // (d,key) = k[key][d]
      v16h kb_i = frag_b_cnk(ki, HD, 0, snt * 16);
      v16h kb_s = frag_b_cnk(ks, HD, 0, snt * 16);
      v8f p1 = {}, p2 = {}, p3 = {};
      p1 = wmma_f16(qa_r, kb_r, p1);
      p2 = wmma_f16(qa_i, kb_i, p2);
      p3 = wmma_f16(qa_s, kb_s, p3);
      int ncol  = snt * 16 + (lane & 15);            // key token
      int mbase = smt * 16 + ((lane & 16) ? 8 : 0);  // query token base
      int kreg  = region9(iw * 8 + (ncol >> 3), jw * 8 + (ncol & 7));
#pragma unroll
      for (int v = 0; v < 8; ++v) {
        int m  = mbase + v;
        int dh = (m >> 3) - (ncol >> 3) + 7;
        int dw = (m & 7)  - (ncol & 7)  + 7;
        int ridx = dh * 15 + dw;
        int qreg = region9(iw * 8 + (m >> 3), jw * 8 + (m & 7));
        float mval = (qreg != kreg) ? -100.0f : 0.0f;
        sre[m * N + ncol] = p1[v] - p2[v] + rpb_r[ridx * NH + hh] + mval;
        sim[m * N + ncol] = p3[v] - p1[v] - p2[v] + rpb_i[ridx * NH + hh];
      }
    }
    __syncthreads();

    // ---- (c) complex softmax per row; write swizzled attn planes ----
    {
      int row = tid >> 2;
      int c0  = (tid & 3) * 16;
      float mx = -1e30f;
#pragma unroll
      for (int j = 0; j < 16; ++j) mx = fmaxf(mx, sre[row * N + c0 + j]);
      mx = fmaxf(mx, __shfl_xor(mx, 1));
      mx = fmaxf(mx, __shfl_xor(mx, 2));
      float er[16], ei[16], ssr = 0.f, ssi = 0.f;
#pragma unroll
      for (int j = 0; j < 16; ++j) {
        float e = __expf(sre[row * N + c0 + j] - mx);
        float sv, cv;
        __sincosf(sim[row * N + c0 + j], &sv, &cv);
        er[j] = e * cv;  ei[j] = e * sv;
        ssr += er[j];    ssi += ei[j];
      }
      ssr += __shfl_xor(ssr, 1); ssr += __shfl_xor(ssr, 2);
      ssi += __shfl_xor(ssi, 1); ssi += __shfl_xor(ssi, 2);
      float inv = 1.0f / (ssr * ssr + ssi * ssi);    // e/s = e*conj(s)/|s|^2
#pragma unroll
      for (int j = 0; j < 16; j += 2) {              // adjacent cols share a dword
        float ar0 = (er[j] * ssr + ei[j] * ssi) * inv;
        float ai0 = (ei[j] * ssr - er[j] * ssi) * inv;
        float ar1 = (er[j+1] * ssr + ei[j+1] * ssi) * inv;
        float ai1 = (ei[j+1] * ssr - er[j+1] * ssi) * inv;
        int a0 = aswz(row, c0 + j, 2);               // even col -> even j slot
        store_h2(afr + a0, ar0, ar1);
        store_h2(afi + a0, ai0, ai1);
        store_h2(afs + a0, ar0 + ai0, ar1 + ai1);
      }
    }
    __syncthreads();

    // ---- (d) O_h = A * v (B-frags from transposed V) ----
    {
      v8f p1 = {}, p2 = {}, p3 = {};
#pragma unroll
      for (int kk = 0; kk < 2; ++kk) {
        int k0 = kk * 32;
        v16h fa_r = frag_a_swz(afr, 2, mt, kk);
        v16h fa_i = frag_a_swz(afi, 2, mt, kk);
        v16h fa_s = frag_a_swz(afs, 2, mt, kk);
        v16h fb_r = frag_b_cnk(vtr, N, k0, nt * 16);  // (tok,d) = vT[d][tok]
        v16h fb_i = frag_b_cnk(vti, N, k0, nt * 16);
        v16h fb_s = frag_b_cnk(vts, N, k0, nt * 16);
        p1 = wmma_f16(fa_r, fb_r, p1);
        p2 = wmma_f16(fa_i, fb_i, p2);
        p3 = wmma_f16(fa_s, fb_s, p3);
      }
      int col  = hh * HD + nt * 16 + (lane & 15);
      int mrow = mt * 16 + ((lane & 16) ? 8 : 0);
#pragma unroll
      for (int v = 0; v < 8; ++v) {
        float cr = p1[v] - p2[v];
        float ci = p3[v] - p1[v] - p2[v];
        int a0 = aswz(mrow + v, col, 8);
        Or[a0] = (_Float16)cr;
        Oi[a0] = (_Float16)ci;
        Os[a0] = (_Float16)(cr + ci);
      }
    }
    __syncthreads();
  }

  // ---- Phase 3: projection (64x256)x(256x256), un-shift, scatter ----
#pragma unroll 1
  for (int t = 0; t < 8; ++t) {
    int p = wid * 8 + t;
    int pmt = p >> 4, pnt = p & 15;
    v8f p1 = {}, p2 = {}, p3 = {};
#pragma unroll
    for (int kk = 0; kk < 8; ++kk) {
      int k0 = kk * 32;
      v16h fa_r = frag_a_swz(Or, 8, pmt, kk);
      v16h fa_i = frag_a_swz(Oi, 8, pmt, kk);
      v16h fa_s = frag_a_swz(Os, 8, pmt, kk);
      v16h fb_r = frag_b_cnk(ws + OP_R, C, k0, pnt * 16);
      v16h fb_i = frag_b_cnk(ws + OP_I, C, k0, pnt * 16);
      v16h fb_s = frag_b_cnk(ws + OP_S, C, k0, pnt * 16);
      p1 = wmma_f16(fa_r, fb_r, p1);
      p2 = wmma_f16(fa_i, fb_i, p2);
      p3 = wmma_f16(fa_s, fb_s, p3);
    }
    int col   = pnt * 16 + (lane & 15);
    int mbase = pmt * 16 + ((lane & 16) ? 8 : 0);
    float br = bproj_r[col], bi = bproj_i[col];
#pragma unroll
    for (int v = 0; v < 8; ++v) {
      int m = mbase + v;
      int h = (iw * 8 + (m >> 3) + SH) & 63;   // window-reverse + roll-back
      int w = (jw * 8 + (m & 7)  + SW) & 63;
      int g = ((b * H + h) * W + w) * C + col;
      out[g]       = p1[v] - p2[v] + br;
      out[IMG + g] = p3[v] - p1[v] - p2[v] + bi;
    }
  }
}

extern "C" void kernel_launch(void* const* d_in, const int* in_sizes, int n_in,
                              void* d_out, int out_size, void* d_ws, size_t ws_size,
                              hipStream_t stream) {
  (void)in_sizes; (void)n_in; (void)out_size; (void)ws_size;
  _Float16* wsh = (_Float16*)d_ws;   // needs 1.5 MB of scratch

  cwa_cvt_weights<<<WQKV / 256, 256, 0, stream>>>(
      (const float*)d_in[2], (const float*)d_in[3],
      (const float*)d_in[6], (const float*)d_in[7], wsh);

  (void)hipFuncSetAttribute(reinterpret_cast<const void*>(cwa_fused_kernel),
                            hipFuncAttributeMaxDynamicSharedMemorySize, LDS_BYTES);
  cwa_fused_kernel<<<B * 64, 256, LDS_BYTES, stream>>>(
      (const float*)d_in[0],  (const float*)d_in[1],   // x r/i
      (const float*)d_in[4],  (const float*)d_in[5],   // b_qkv r/i
      (const float*)d_in[8],  (const float*)d_in[9],   // b_proj r/i
      (const float*)d_in[10], (const float*)d_in[11],  // rpb r/i
      wsh, (float*)d_out);
}
